// AdaptiveMLP_27874337751265
// MI455X (gfx1250) — matrix operand order repaired and checked by reading, not compiled, not run on hardware
//
#include <hip/hip_runtime.h>
#include <hip/hip_bf16.h>

// ---------------- problem constants ----------------
#define IMG_H   448
#define IMG_W   448
#define HWPIX   (448*448)
#define NBATCH  64
#define NTOK    256
#define DDIM    768
#define HDIM    768
#define NBINS   36
#define TWO_PI  6.283185307179586f

typedef __attribute__((ext_vector_type(16))) _Float16 v16h;
typedef __attribute__((ext_vector_type(8)))  float    v8f;
typedef __attribute__((ext_vector_type(4)))  unsigned int v4u;
typedef __attribute__((ext_vector_type(4)))  int      v4i;
typedef __attribute__((ext_vector_type(8)))  int      v8i;

struct alignas(16) h8 { _Float16 h[8]; };

// D = A(16x32 f16) x B(32x16 f16) + C(16x16 f32)  -> v_wmma_f32_16x16x32_f16
#define WMMA_F16(a, b, c) \
  __builtin_amdgcn_wmma_f32_16x16x32_f16(false, (a), false, (b), (short)0, (c), false, false)

__device__ __forceinline__ v16h load_two_chunks(const _Float16* p0, const _Float16* p1) {
  v16h r;
  float4* d = reinterpret_cast<float4*>(&r);
  d[0] = *reinterpret_cast<const float4*>(p0);
  d[1] = *reinterpret_cast<const float4*>(p1);
  return r;
}

// A fragment, 16x32 f16, row-major A[lda] (contiguous along K).
// Lane L: M = m0 + (L&15); elems 0..7 <- K=k0+(L>>4)*8 ; elems 8..15 <- +16
__device__ __forceinline__ v16h load_a_frag(const _Float16* __restrict__ A, int lda,
                                            int m0, int k0, int lane) {
  const int m = m0 + (lane & 15);
  const _Float16* p = A + (size_t)m * lda + k0 + (lane >> 4) * 8;
  return load_two_chunks(p, p + 16);
}

// B fragment, 32x16 f16: B(k,n) stored as Bt[n][k] row-major (contiguous along K).
// Lane L: N = n0 + (L&15); elems 0..7 <- K=k0+(L>>4)*8 ; elems 8..15 <- +16
__device__ __forceinline__ v16h load_b_frag(const _Float16* __restrict__ Bt, int ldb,
                                            int n0, int k0, int lane) {
  const int n = n0 + (lane & 15);
  const _Float16* p = Bt + (size_t)n * ldb + k0 + (lane >> 4) * 8;
  return load_two_chunks(p, p + 16);
}

__device__ __forceinline__ float rand01(unsigned x, unsigned seed) {
  x = (x ^ seed) * 0x9E3779B1u;
  x ^= x >> 16; x *= 0x85EBCA6Bu;
  x ^= x >> 13; x *= 0xC2B2AE35u;
  x ^= x >> 16;
  return (float)(x >> 8) * (1.0f / 16777216.0f);
}

// ---- TDM: stage a contiguous `nelem` f16 slab (global -> LDS offset 0) ----
// D# per CDNA5 ISA ch.8: group0 {count=1, lds_addr, global_addr, type=2},
// group1 {data_size=2B, dim0=tile0=nelem, dim1=tile1=1, stride0=nelem}, grp2/3 NULL.
__device__ __forceinline__ void tdm_load_slab_f16(const _Float16* gsrc, unsigned nelem) {
#if __has_builtin(__builtin_amdgcn_tensor_load_to_lds)
  const unsigned lo = (unsigned)(size_t)gsrc;
  const unsigned hi = (unsigned)((size_t)gsrc >> 32);
  v4u g0;
  g0[0] = 1u;                                   // count=1 (valid descriptor)
  g0[1] = 0u;                                   // lds_addr = 0 (slab is first LDS object)
  g0[2] = lo;                                   // global_addr[31:0]
  g0[3] = (hi & 0x01FFFFFFu) | (2u << 30);      // global_addr[56:32] | type=2
  v8i g1;
  g1[0] = 0x00010000;                           // wg_mask=0, data_size=1 (2 bytes)
  g1[1] = (int)((nelem & 0xFFFFu) << 16);       // tensor_dim0[15:0]
  g1[2] = (int)(((nelem >> 16) & 0xFFFFu) | (1u << 16));  // dim0[31:16] | tensor_dim1=1
  g1[3] = (int)((nelem & 0xFFFFu) << 16);       // tensor_dim1[31:16]=0 | tile_dim0
  g1[4] = 1;                                    // tile_dim1=1, tile_dim2=0
  g1[5] = (int)nelem;                           // tensor_dim0_stride[31:0]
  g1[6] = 0;
  g1[7] = 0;
  v4i z4 = {};
#if defined(__clang_major__) && (__clang_major__ >= 23)
  v8i z8 = {};
  __builtin_amdgcn_tensor_load_to_lds(g0, g1, z4, z4, z8, 0);
#else
  __builtin_amdgcn_tensor_load_to_lds(g0, g1, z4, z4, 0);
#endif
  __builtin_amdgcn_s_wait_tensorcnt(0);
#endif
}

// The TDM builtin writes LDS invisibly to the compiler. Escape the slab pointer
// through an opaque asm with a memory clobber (input-only operand keeps the
// pointer value - and thus LDS addrspace inference / ds_load codegen - intact).
__device__ __forceinline__ void tdm_publish(_Float16* lds) {
  asm volatile("" :: "v"(lds) : "memory");
}

// ---------------- 1) grayscale + mean-subtract + Hann window -> f16 ----------------
__global__ void k_gray(const float* __restrict__ img, _Float16* __restrict__ gray) {
  const int b = blockIdx.x;
  const int tid = threadIdx.x;
  const float* base = img + (size_t)b * 3 * HWPIX;
  float sum = 0.f;
  for (int i = tid; i < HWPIX; i += 256) {
    float r  = base[i], g = base[HWPIX + i], bl = base[2 * HWPIX + i];
    float gy = (r * 0.229f + 0.485f) * 0.2989f +
               (g * 0.224f + 0.456f) * 0.587f  +
               (bl * 0.225f + 0.406f) * 0.114f;
    sum += gy;
  }
  __shared__ float red[256];
  red[tid] = sum;
  __syncthreads();
  for (int s = 128; s > 0; s >>= 1) { if (tid < s) red[tid] += red[tid + s]; __syncthreads(); }
  const float mean = red[0] * (1.0f / HWPIX);
  _Float16* gout = gray + (size_t)b * HWPIX;
  for (int i = tid; i < HWPIX; i += 256) {
    float r  = base[i], g = base[HWPIX + i], bl = base[2 * HWPIX + i];
    float gy = (r * 0.229f + 0.485f) * 0.2989f +
               (g * 0.224f + 0.456f) * 0.587f  +
               (bl * 0.225f + 0.406f) * 0.114f;
    int y = i / IMG_W, x = i - y * IMG_W;
    float hy = 0.5f - 0.5f * cosf((TWO_PI / IMG_H) * (float)y);
    float hx = 0.5f - 0.5f * cosf((TWO_PI / IMG_W) * (float)x);
    gout[i] = (_Float16)((gy - mean) * hy * hx);
  }
}

// ---------------- 2) DFT matrices (symmetric) + negated sin + bin map ----------------
__global__ void k_init_dft(_Float16* __restrict__ cosM, _Float16* __restrict__ sinM,
                           _Float16* __restrict__ nsinM, signed char* __restrict__ binmap) {
  const int i = blockIdx.x * 256 + threadIdx.x;   // exactly 784*256 = HWPIX
  const int u = i / IMG_W, x = i - u * IMG_W;
  const int m = (int)(((long long)u * (long long)x) % IMG_W);
  const float ang = (TWO_PI / IMG_W) * (float)m;
  const float cv = cosf(ang), sv = sinf(ang);
  cosM[i]  = (_Float16)cv;
  sinM[i]  = (_Float16)sv;
  nsinM[i] = (_Float16)(-sv);
  const int si = (u + IMG_H / 2) % IMG_H;
  const int sj = (x + IMG_W / 2) % IMG_W;
  const float yy = (float)si - 223.5f;
  const float xx = (float)sj - 223.5f;
  const float r = sqrtf(xx * xx + yy * yy);
  const float rmax = sqrtf(2.0f) * 223.5f;
  float theta = atan2f(yy, xx);
  theta = fmodf(theta + TWO_PI, TWO_PI);
  int bin = (int)(theta / (TWO_PI / NBINS));
  bin = bin < 0 ? 0 : (bin > NBINS - 1 ? NBINS - 1 : bin);
  const bool mask = (r >= 0.05f * rmax) && (r <= 0.95f * rmax);
  binmap[i] = mask ? (signed char)bin : (signed char)-1;
}

__global__ void k_zero_energy(float* __restrict__ energy) {
  const int i = blockIdx.x * 256 + threadIdx.x;
  if (i < NBATCH * NBINS) energy[i] = 0.f;
}

// ------- 3) row DFT: Re1 = G*C, Im1 = G*(-S)  (stored transposed [v][y]) -------
__global__ void k_dft_row(const _Float16* __restrict__ gray,
                          const _Float16* __restrict__ cosM,
                          const _Float16* __restrict__ nsinM,
                          _Float16* __restrict__ re1t, _Float16* __restrict__ im1t) {
  const int b = blockIdx.y;
  const int lane = threadIdx.x & 31, wave = threadIdx.x >> 5;   // 4 waves/block
  const int t  = blockIdx.x * 4 + wave;                          // 0..195
  const int m0 = (t / 14) * 32;
  const int n0 = (t % 14) * 32;
  const _Float16* A = gray + (size_t)b * HWPIX;
  v8f accRe[4] = {{}, {}, {}, {}};
  v8f accIm[4] = {{}, {}, {}, {}};
  for (int k0 = 0; k0 < IMG_W; k0 += 32) {
    v16h a0 = load_a_frag(A, IMG_W, m0,      k0, lane);
    v16h a1 = load_a_frag(A, IMG_W, m0 + 16, k0, lane);
    v16h c0 = load_b_frag(cosM,  IMG_W, n0,      k0, lane);  // symmetric: Bt == C
    v16h c1 = load_b_frag(cosM,  IMG_W, n0 + 16, k0, lane);
    v16h s0 = load_b_frag(nsinM, IMG_W, n0,      k0, lane);
    v16h s1 = load_b_frag(nsinM, IMG_W, n0 + 16, k0, lane);
    accRe[0] = WMMA_F16(a0, c0, accRe[0]);
    accRe[1] = WMMA_F16(a0, c1, accRe[1]);
    accRe[2] = WMMA_F16(a1, c0, accRe[2]);
    accRe[3] = WMMA_F16(a1, c1, accRe[3]);
    accIm[0] = WMMA_F16(a0, s0, accIm[0]);
    accIm[1] = WMMA_F16(a0, s1, accIm[1]);
    accIm[2] = WMMA_F16(a1, s0, accIm[2]);
    accIm[3] = WMMA_F16(a1, s1, accIm[3]);
  }
  const int half = lane >> 4;
#pragma unroll
  for (int ti = 0; ti < 4; ++ti) {
    const int mi = m0 + (ti >> 1) * 16;
    const int v  = n0 + (ti & 1) * 16 + (lane & 15);
    h8 pr, pi;
#pragma unroll
    for (int r = 0; r < 8; ++r) { pr.h[r] = (_Float16)accRe[ti][r]; pi.h[r] = (_Float16)accIm[ti][r]; }
    const size_t o = (size_t)b * HWPIX + (size_t)v * IMG_H + mi + half * 8;  // [v][y]
    *reinterpret_cast<float4*>(re1t + o) = *reinterpret_cast<float4*>(&pr);
    *reinterpret_cast<float4*>(im1t + o) = *reinterpret_cast<float4*>(&pi);
  }
}

// ------ 4) column DFT + |F|^2 + angular-bin accumulation (LDS bins -> global) -------
__global__ void k_dft_col(const _Float16* __restrict__ cosM,
                          const _Float16* __restrict__ sinM,
                          const _Float16* __restrict__ nsinM,
                          const _Float16* __restrict__ re1t,
                          const _Float16* __restrict__ im1t,
                          const signed char* __restrict__ binmap,
                          float* __restrict__ energy) {
  __shared__ float bins[NBINS];
  const int tid = threadIdx.x;
  if (tid < NBINS) bins[tid] = 0.f;
  __syncthreads();

  const int b = blockIdx.y;
  const int lane = tid & 31, wave = tid >> 5;   // 4 waves/block
  const int t  = blockIdx.x * 4 + wave;
  const int m0 = (t / 14) * 32;
  const int n0 = (t % 14) * 32;
  const _Float16* Re = re1t + (size_t)b * HWPIX;   // [v][y]
  const _Float16* Im = im1t + (size_t)b * HWPIX;
  v8f aRe[4] = {{}, {}, {}, {}};
  v8f aIm[4] = {{}, {}, {}, {}};
  for (int k0 = 0; k0 < IMG_H; k0 += 32) {
    v16h c0  = load_a_frag(cosM,  IMG_H, m0,      k0, lane);
    v16h c1  = load_a_frag(cosM,  IMG_H, m0 + 16, k0, lane);
    v16h s0  = load_a_frag(sinM,  IMG_H, m0,      k0, lane);
    v16h s1  = load_a_frag(sinM,  IMG_H, m0 + 16, k0, lane);
    v16h ns0 = load_a_frag(nsinM, IMG_H, m0,      k0, lane);
    v16h ns1 = load_a_frag(nsinM, IMG_H, m0 + 16, k0, lane);
    v16h zr0 = load_b_frag(Re, IMG_H, n0,      k0, lane);
    v16h zr1 = load_b_frag(Re, IMG_H, n0 + 16, k0, lane);
    v16h zi0 = load_b_frag(Im, IMG_H, n0,      k0, lane);
    v16h zi1 = load_b_frag(Im, IMG_H, n0 + 16, k0, lane);
    aRe[0] = WMMA_F16(c0, zr0, aRe[0]);  aRe[0] = WMMA_F16(s0, zi0, aRe[0]);
    aRe[1] = WMMA_F16(c0, zr1, aRe[1]);  aRe[1] = WMMA_F16(s0, zi1, aRe[1]);
    aRe[2] = WMMA_F16(c1, zr0, aRe[2]);  aRe[2] = WMMA_F16(s1, zi0, aRe[2]);
    aRe[3] = WMMA_F16(c1, zr1, aRe[3]);  aRe[3] = WMMA_F16(s1, zi1, aRe[3]);
    aIm[0] = WMMA_F16(c0, zi0, aIm[0]);  aIm[0] = WMMA_F16(ns0, zr0, aIm[0]);
    aIm[1] = WMMA_F16(c0, zi1, aIm[1]);  aIm[1] = WMMA_F16(ns0, zr1, aIm[1]);
    aIm[2] = WMMA_F16(c1, zi0, aIm[2]);  aIm[2] = WMMA_F16(ns1, zr0, aIm[2]);
    aIm[3] = WMMA_F16(c1, zi1, aIm[3]);  aIm[3] = WMMA_F16(ns1, zr1, aIm[3]);
  }
  const int half = lane >> 4;
#pragma unroll
  for (int ti = 0; ti < 4; ++ti) {
    const int mi = m0 + (ti >> 1) * 16;
    const int v  = n0 + (ti & 1) * 16 + (lane & 15);
#pragma unroll
    for (int r = 0; r < 8; ++r) {
      const int u = mi + r + half * 8;
      const float re = aRe[ti][r], im = aIm[ti][r];
      const float P = re * re + im * im;
      const int bin = (int)binmap[u * IMG_W + v];
      if (bin >= 0) atomicAdd(&bins[bin], P);
    }
  }
  __syncthreads();
  if (tid < NBINS) atomicAdd(&energy[b * NBINS + tid], bins[tid]);
}

// ---------------- 5) angular entropy -> APE -> dropout probability ----------------
__global__ void k_probs(const float* __restrict__ energy, float* __restrict__ probs) {
  const int b = threadIdx.x;
  if (b >= NBATCH) return;
  const float* e = energy + b * NBINS;
  float total = 0.f;
  for (int j = 0; j < NBINS; ++j) total += e[j];
  float Ht = 0.f;
  for (int j = 0; j < NBINS; ++j) {
    float p = e[j] / (total + 1e-12f);
    Ht -= p * logf(p + 1e-12f);
  }
  float ape = Ht / logf((float)NBINS);
  ape = fminf(fmaxf(ape, 0.f), 1.f);
  if (total <= 1e-12f) ape = 0.f;
  const float adj = tanhf((0.78f - ape) * 4.0f);
  float drop = (adj >= 0.f) ? (0.3f + adj * (0.6f - 0.3f))
                            : (0.3f + adj * (0.3f - 0.1f));
  probs[b] = fminf(fmaxf(drop, 0.1f), 0.6f);
}

// ---------------- 6) dropout on x + convert to f16 ----------------
__global__ void k_drop_x(const float* __restrict__ x, const float* __restrict__ probs,
                         _Float16* __restrict__ xh) {
  const unsigned i = blockIdx.x * 256u + threadIdx.x;   // exactly B*N*D
  const float p = probs[i / (NTOK * DDIM)];
  const float rn = rand01(i, 0xC0FFEE01u);
  const float keep = (rn < 1.f - p) ? 1.f / (1.f - p + 1e-8f) : 0.f;
  xh[i] = (_Float16)(x[i] * keep);
}

__global__ void k_wconv(const float* __restrict__ in, _Float16* __restrict__ out, int n) {
  const int i = blockIdx.x * 256 + threadIdx.x;
  if (i < n) out[i] = (_Float16)in[i];
}

// ---- cooperative fallback copy (only used if TDM builtin is unavailable) ----
__device__ __forceinline__ void coop_load_slab(_Float16* lds, const _Float16* gsrc,
                                               int nelem, int tid, int nthreads) {
#if !__has_builtin(__builtin_amdgcn_tensor_load_to_lds)
  const int nvec = nelem / 8;                       // float4-sized chunks
  for (int i = tid; i < nvec; i += nthreads)
    reinterpret_cast<float4*>(lds)[i] = reinterpret_cast<const float4*>(gsrc)[i];
#endif
}

// ---------------- 7) GEMM1: h = gelu(x @ w1^T) with dropout, f16 out ----------------
// All 8 waves of a block share one 32-row weight slab, staged in LDS by the TDM.
// Per wave: 4x2 register blocking (4 global A loads + 2 LDS B loads -> 8 WMMAs).
__global__ void k_gemm1(const _Float16* __restrict__ xh, const _Float16* __restrict__ w1h,
                        const float* __restrict__ probs, _Float16* __restrict__ hbuf) {
  __shared__ alignas(16) _Float16 wslab[32 * DDIM];   // 48 KB, first LDS object
  const int lane = threadIdx.x & 31, wave = threadIdx.x >> 5;
  const int tn = blockIdx.x % (HDIM / 32);
  const int tm = (blockIdx.x / (HDIM / 32)) * 8 + wave;
  const int m0 = tm * 64, n0 = tn * 32;

  if (wave == 0) tdm_load_slab_f16(w1h + (size_t)n0 * DDIM, 32 * DDIM);  // TDM, wave0 only
  coop_load_slab(wslab, w1h + (size_t)n0 * DDIM, 32 * DDIM, threadIdx.x, 256);
  tdm_publish(wslab);               // compiler: the TDM wrote this slab
  __syncthreads();

  v8f acc[4][2] = {{{}, {}}, {{}, {}}, {{}, {}}, {{}, {}}};
#pragma unroll
  for (int k0 = 0; k0 < DDIM; k0 += 32) {
    if (k0 + 32 < DDIM)
      __builtin_prefetch(xh + (size_t)(m0 + (lane & 15)) * DDIM + k0 + 32, 0, 1);
    v16h b0 = load_b_frag(wslab, DDIM, 0,  k0, lane);   // ds_load from staged slab
    v16h b1 = load_b_frag(wslab, DDIM, 16, k0, lane);
#pragma unroll
    for (int i = 0; i < 4; ++i) {
      v16h a = load_a_frag(xh, DDIM, m0 + i * 16, k0, lane);
      acc[i][0] = WMMA_F16(a, b0, acc[i][0]);
      acc[i][1] = WMMA_F16(a, b1, acc[i][1]);
    }
  }
  const int half = lane >> 4;
#pragma unroll
  for (int i = 0; i < 4; ++i)
#pragma unroll
    for (int j = 0; j < 2; ++j) {
      const int n = n0 + j * 16 + (lane & 15);
#pragma unroll
      for (int r = 0; r < 8; ++r) {
        const int m = m0 + i * 16 + r + half * 8;
        float v = acc[i][j][r];
        v = 0.5f * v * (1.f + erff(v * 0.70710678f));           // exact GELU
        const float p = probs[m / NTOK];
        const float rn = rand01((unsigned)(m * HDIM + n), 0xC0FFEE02u);
        const float keep = (rn < 1.f - p) ? 1.f / (1.f - p + 1e-8f) : 0.f;
        hbuf[(size_t)m * HDIM + n] = (_Float16)(v * keep);
      }
    }
}

// ---------------- 8) GEMM2: y = h @ w2^T with dropout, f32 out ----------------
__global__ void k_gemm2(const _Float16* __restrict__ hbuf, const _Float16* __restrict__ w2h,
                        const float* __restrict__ probs, float* __restrict__ out) {
  __shared__ alignas(16) _Float16 wslab[32 * HDIM];   // 48 KB, first LDS object
  const int lane = threadIdx.x & 31, wave = threadIdx.x >> 5;
  const int tn = blockIdx.x % (DDIM / 32);
  const int tm = (blockIdx.x / (DDIM / 32)) * 8 + wave;
  const int m0 = tm * 64, n0 = tn * 32;

  if (wave == 0) tdm_load_slab_f16(w2h + (size_t)n0 * HDIM, 32 * HDIM);
  coop_load_slab(wslab, w2h + (size_t)n0 * HDIM, 32 * HDIM, threadIdx.x, 256);
  tdm_publish(wslab);
  __syncthreads();

  v8f acc[4][2] = {{{}, {}}, {{}, {}}, {{}, {}}, {{}, {}}};
#pragma unroll
  for (int k0 = 0; k0 < HDIM; k0 += 32) {
    if (k0 + 32 < HDIM)
      __builtin_prefetch(hbuf + (size_t)(m0 + (lane & 15)) * HDIM + k0 + 32, 0, 1);
    v16h b0 = load_b_frag(wslab, HDIM, 0,  k0, lane);
    v16h b1 = load_b_frag(wslab, HDIM, 16, k0, lane);
#pragma unroll
    for (int i = 0; i < 4; ++i) {
      v16h a = load_a_frag(hbuf, HDIM, m0 + i * 16, k0, lane);
      acc[i][0] = WMMA_F16(a, b0, acc[i][0]);
      acc[i][1] = WMMA_F16(a, b1, acc[i][1]);
    }
  }
  const int half = lane >> 4;
#pragma unroll
  for (int i = 0; i < 4; ++i)
#pragma unroll
    for (int j = 0; j < 2; ++j) {
      const int n = n0 + j * 16 + (lane & 15);
#pragma unroll
      for (int r = 0; r < 8; ++r) {
        const int m = m0 + i * 16 + r + half * 8;
        float v = acc[i][j][r];
        const float p = probs[m / NTOK];
        const float rn = rand01((unsigned)(m * DDIM + n), 0xC0FFEE03u);
        const float keep = (rn < 1.f - p) ? 1.f / (1.f - p + 1e-8f) : 0.f;
        out[(size_t)m * DDIM + n] = v * keep;
      }
    }
}

// =====================================================================
extern "C" void kernel_launch(void* const* d_in, const int* in_sizes, int n_in,
                              void* d_out, int out_size, void* d_ws, size_t ws_size,
                              hipStream_t stream) {
  const float* images = (const float*)d_in[0];
  const float* x      = (const float*)d_in[1];
  const float* w1     = (const float*)d_in[2];
  const float* w2     = (const float*)d_in[3];
  float* out = (float*)d_out;

  char* ws = (char*)d_ws;
  size_t off = 0;
  auto take = [&](size_t bytes) -> char* {
    char* p = ws + off;
    off += (bytes + 255) & ~(size_t)255;
    return p;
  };
  _Float16* gray   = (_Float16*)take((size_t)NBATCH * HWPIX * 2);  // reused as xh
  _Float16* re1t   = (_Float16*)take((size_t)NBATCH * HWPIX * 2);  // reused as hbuf
  _Float16* im1t   = (_Float16*)take((size_t)NBATCH * HWPIX * 2);
  _Float16* cosM   = (_Float16*)take((size_t)HWPIX * 2);
  _Float16* sinM   = (_Float16*)take((size_t)HWPIX * 2);
  _Float16* nsinM  = (_Float16*)take((size_t)HWPIX * 2);
  signed char* binmap = (signed char*)take((size_t)HWPIX);
  float*    energy = (float*)   take((size_t)NBATCH * NBINS * 4);
  float*    probs  = (float*)   take((size_t)NBATCH * 4);
  _Float16* w1h    = (_Float16*)take((size_t)HDIM * DDIM * 2);
  _Float16* w2h    = (_Float16*)take((size_t)DDIM * HDIM * 2);
  _Float16* xh   = gray;   // gray dead after k_dft_row
  _Float16* hbuf = re1t;   // re1t dead after k_dft_col

  // weights to f16 (independent, launch first)
  k_wconv<<<(HDIM * DDIM + 255) / 256, 256, 0, stream>>>(w1, w1h, HDIM * DDIM);
  k_wconv<<<(DDIM * HDIM + 255) / 256, 256, 0, stream>>>(w2, w2h, DDIM * HDIM);

  // APE pipeline
  k_gray<<<NBATCH, 256, 0, stream>>>(images, gray);
  k_init_dft<<<HWPIX / 256, 256, 0, stream>>>(cosM, sinM, nsinM, binmap);
  k_zero_energy<<<(NBATCH * NBINS + 255) / 256, 256, 0, stream>>>(energy);
  {
    dim3 g((14 * 14) / 4, NBATCH);  // 4 waves/block, one 32x32 macro-tile per wave
    k_dft_row<<<g, 128, 0, stream>>>(gray, cosM, nsinM, re1t, im1t);
    k_dft_col<<<g, 128, 0, stream>>>(cosM, sinM, nsinM, re1t, im1t, binmap, energy);
  }
  k_probs<<<1, 64, 0, stream>>>(energy, probs);

  // MLP pipeline
  k_drop_x<<<(NBATCH * NTOK * DDIM) / 256, 256, 0, stream>>>(x, probs, xh);
  const int M = NBATCH * NTOK;
  k_gemm1<<<(M / 64) * (HDIM / 32) / 8, 256, 0, stream>>>(xh, w1h, probs, hbuf);
  k_gemm2<<<(M / 64) * (DDIM / 32) / 8, 256, 0, stream>>>(hbuf, w2h, probs, out);
}
